// CrossAtt_3075196584295
// MI455X (gfx1250) — compile-verified, matrix operand order checked by reading, not run on hardware
//
#include <hip/hip_runtime.h>
#include <hip/hip_bf16.h>
#include <math.h>
#include <stdint.h>

// ---------------------------------------------------------------------------
// CDNA5 (gfx1250, wave32) cross-attention stack.
// GEMMs + attention scores: v_wmma_f32_16x16x32_f16; f16 operands staged via
// LDS (global_load_async_to_lds_b128) or read directly; f32 accumulate.
// ---------------------------------------------------------------------------

typedef __attribute__((ext_vector_type(16))) _Float16 v16h;
typedef __attribute__((ext_vector_type(8)))  _Float16 v8h;
typedef __attribute__((ext_vector_type(4)))  _Float16 v4h;
typedef __attribute__((ext_vector_type(8)))  float    v8f;
typedef __attribute__((ext_vector_type(4)))  int      v4i;

#define LAYERS 4
#define DMODEL 256
#define NHEAD  8
#define HDIM   32
#define BATCH  2
#define SEQQ   1024
#define SEQK   1024
#define FFDIM  768
#define NTOK   (BATCH * SEQQ)      // 2048 rows (== B*SK)
#define EPSLN  1e-5f

#define ACT_NONE 0
#define ACT_MISH 1

#if defined(__gfx1250__) && __has_builtin(__builtin_amdgcn_global_load_async_to_lds_b128)
#define HAS_ASYNC_LDS 1
#else
#define HAS_ASYNC_LDS 0
#endif

#if HAS_ASYNC_LDS
typedef __attribute__((address_space(1))) v4i as1_v4i;   // global v4i
typedef __attribute__((address_space(3))) v4i as3_v4i;   // LDS v4i
__device__ __forceinline__ void async_copy_b128(const _Float16* gsrc, _Float16* lds_dst) {
  __builtin_amdgcn_global_load_async_to_lds_b128(
      (as1_v4i*)(uintptr_t)gsrc, (as3_v4i*)(uintptr_t)lds_dst, 0, 0);
}
__device__ __forceinline__ void wait_async_zero() {
#if __has_builtin(__builtin_amdgcn_s_wait_asynccnt)
  __builtin_amdgcn_s_wait_asynccnt(0);
#else
  asm volatile("s_wait_asynccnt 0x0" ::: "memory");
#endif
}
#endif

__device__ __forceinline__ float mishf(float x) {
  float sp = log1pf(expf(x));
  return x * tanhf(sp);
}

__device__ __forceinline__ v16h cat16(v8h lo, v8h hi) {
  return __builtin_shufflevector(lo, hi, 0,1,2,3,4,5,6,7,8,9,10,11,12,13,14,15);
}

// ---------------------------------------------------------------------------
// WMMA GEMM:  C[M,N] = act(A[M,K] @ W[K,N] + bias[N]) (+ residual)
// A, W are f16. Block = 256 threads = 8 waves; block tile 128(M) x 64(N);
// wave grid 4x2, each wave computes 32x32 via 4 accumulators -> 4 WMMA / step.
// ---------------------------------------------------------------------------
template <int K>
__global__ __launch_bounds__(256) void gemm_wmma_kernel(
    const _Float16* __restrict__ A, const _Float16* __restrict__ W,
    const float* __restrict__ bias, const float* __restrict__ residual,
    float* __restrict__ C, _Float16* __restrict__ C16, int N, int act)
{
  constexpr int LDA = 40;               // padded row stride (halves)
  __shared__ _Float16 As[128][LDA];     // 128 rows x 32 k
  __shared__ _Float16 BsT[64][LDA];     // 64 cols  x 32 k (transposed tile)

  const int tid  = threadIdx.x;
  const int lane = tid & 31;
  const int wave = tid >> 5;
  const int wm   = wave >> 1;
  const int wn   = wave & 1;
  const int m0   = blockIdx.x * 128;
  const int n0   = blockIdx.y * 64;

  const int frag_rc = lane & 15;
  const int khalf   = (lane >> 4) * 8;

  v8f acc[2][2] = {};

  for (int k0 = 0; k0 < K; k0 += 32) {
#pragma unroll
    for (int it = 0; it < 2; ++it) {
      const int e   = tid + it * 256;
      const int r   = e >> 2;
      const int seg = e & 3;
      const _Float16* src = A + (size_t)(m0 + r) * K + k0 + seg * 8;
      _Float16*       dst = &As[r][seg * 8];
#if HAS_ASYNC_LDS
      async_copy_b128(src, dst);
#else
      *(uint4*)dst = *(const uint4*)src;
#endif
    }
    {
      const int kk  = tid >> 3;
      const int seg = tid & 7;
      const v8h wv = *(const v8h*)(W + (size_t)(k0 + kk) * N + n0 + seg * 8);
#pragma unroll
      for (int j = 0; j < 8; ++j) BsT[seg * 8 + j][kk] = wv[j];
    }
#if HAS_ASYNC_LDS
    wait_async_zero();
#endif
    __syncthreads();

    const v16h af0 = cat16(*(const v8h*)&As[wm * 32 + frag_rc][khalf],
                           *(const v8h*)&As[wm * 32 + frag_rc][16 + khalf]);
    const v16h af1 = cat16(*(const v8h*)&As[wm * 32 + 16 + frag_rc][khalf],
                           *(const v8h*)&As[wm * 32 + 16 + frag_rc][16 + khalf]);
    const v16h bf0 = cat16(*(const v8h*)&BsT[wn * 32 + frag_rc][khalf],
                           *(const v8h*)&BsT[wn * 32 + frag_rc][16 + khalf]);
    const v16h bf1 = cat16(*(const v8h*)&BsT[wn * 32 + 16 + frag_rc][khalf],
                           *(const v8h*)&BsT[wn * 32 + 16 + frag_rc][16 + khalf]);

    acc[0][0] = __builtin_amdgcn_wmma_f32_16x16x32_f16(false, af0, false, bf0, (short)0, acc[0][0], false, false);
    acc[0][1] = __builtin_amdgcn_wmma_f32_16x16x32_f16(false, af0, false, bf1, (short)0, acc[0][1], false, false);
    acc[1][0] = __builtin_amdgcn_wmma_f32_16x16x32_f16(false, af1, false, bf0, (short)0, acc[1][0], false, false);
    acc[1][1] = __builtin_amdgcn_wmma_f32_16x16x32_f16(false, af1, false, bf1, (short)0, acc[1][1], false, false);
    __syncthreads();
  }

  const int rsel = (lane >> 4) ? 8 : 0;
#pragma unroll
  for (int sm = 0; sm < 2; ++sm) {
#pragma unroll
    for (int sn = 0; sn < 2; ++sn) {
      const int cc    = n0 + wn * 32 + sn * 16 + (lane & 15);
      const int rbase = m0 + wm * 32 + sm * 16 + rsel;
      const float bvv = bias ? bias[cc] : 0.f;
#pragma unroll
      for (int r = 0; r < 8; ++r) {
        float val = acc[sm][sn][r] + bvv;
        if (act == ACT_MISH) val = mishf(val);
        const size_t idx = (size_t)(rbase + r) * N + cc;
        if (residual) val += residual[idx];
        if (C)   C[idx]   = val;
        if (C16) C16[idx] = (_Float16)val;
      }
    }
  }
}

// ---------------------------------------------------------------------------
__global__ __launch_bounds__(256) void cvt_f16_kernel(
    const float* __restrict__ src, _Float16* __restrict__ dst, int n4)
{
  int i = blockIdx.x * 256 + threadIdx.x;
  if (i < n4) {
    const float4 f = ((const float4*)src)[i];
    v4h h;
    h[0] = (_Float16)f.x; h[1] = (_Float16)f.y;
    h[2] = (_Float16)f.z; h[3] = (_Float16)f.w;
    ((v4h*)dst)[i] = h;
  }
}

// ---------------------------------------------------------------------------
// LayerNorm: out = (add?add:0) + ln(in)*g + b ; optional f16 view out16.
// ---------------------------------------------------------------------------
__global__ __launch_bounds__(256) void ln_add_kernel(
    const float* __restrict__ in, const float* __restrict__ g,
    const float* __restrict__ b, const float* __restrict__ add,
    float* __restrict__ out, _Float16* __restrict__ out16)
{
  __shared__ float red[256];
  __shared__ float s_mean, s_rstd;
  const int tid = threadIdx.x;
  const size_t rowbase = (size_t)blockIdx.x * DMODEL;

  float x = in[rowbase + tid];
  red[tid] = x;
  __syncthreads();
  for (int off = 128; off > 0; off >>= 1) {
    if (tid < off) red[tid] += red[tid + off];
    __syncthreads();
  }
  if (tid == 0) s_mean = red[0] * (1.f / DMODEL);
  __syncthreads();
  float dxm = x - s_mean;
  red[tid] = dxm * dxm;
  __syncthreads();
  for (int off = 128; off > 0; off >>= 1) {
    if (tid < off) red[tid] += red[tid + off];
    __syncthreads();
  }
  if (tid == 0) s_rstd = rsqrtf(red[0] * (1.f / DMODEL) + EPSLN);
  __syncthreads();
  float y = dxm * s_rstd * g[tid] + b[tid];
  if (add) y += add[rowbase + tid];
  if (out)   out[rowbase + tid]   = y;
  if (out16) out16[rowbase + tid] = (_Float16)y;
}

// ---------------------------------------------------------------------------
// WMMA attention with exact alpha=1.5 entmax.
// One block per (b, h, 16-query tile): grid = B*H*SQ/16 = 1024 blocks.
// Scores: S(16x1024) = Qtile(16x32) @ K^T via 64 wmma tiles (8 per wave).
//   A frag = Q rows (2 contiguous b128 per lane); B frag of K^T tile is
//   per-lane contiguous in row-major K (lane = key, elems = head dims):
//   no LDS staging or transpose needed.
// Then entmax15 per row (sort/scan in LDS), then attn @ V (f32 VALU,
// one V stream feeding two rows per thread).
// Dynamic LDS: 16x1024 f32 scores + sort buffers (~84 KB; CDNA5 WGP has 320KB).
// ---------------------------------------------------------------------------
__global__ __launch_bounds__(256) void attn_entmax_wmma_kernel(
    const _Float16* __restrict__ qh, const _Float16* __restrict__ kh,
    const float* __restrict__ v, _Float16* __restrict__ out)
{
  extern __shared__ float smem[];
  float* Ss   = smem;                 // [16][1024]
  float* sbuf = Ss + 16 * SEQK;       // [1024]
  float* cs1  = sbuf + SEQK;          // [1024]
  float* cs2  = cs1 + SEQK;           // [1024]
  float* taub = cs2 + SEQK;           // [1024]
  float* red1 = taub + SEQK;          // [256]
  float* red2 = red1 + 256;           // [256]
  __shared__ int s_support;

  const int tid  = threadIdx.x;
  const int lane = tid & 31;
  const int wave = tid >> 5;
  const int blk  = blockIdx.x;
  const int qt   = blk & (SEQQ / 16 - 1);          // 0..63
  const int h    = (blk >> 6) & (NHEAD - 1);
  const int b    = blk >> 9;
  const int qi0  = qt * 16;

  const int fr    = lane & 15;
  const int khalf = (lane >> 4) * 8;
  const int rsel  = (lane >> 4) ? 8 : 0;
  const float scale = 0.5f * 0.17677669529663689f;   // entmax 0.5 / sqrt(32)

  // ---- scores via WMMA: A fragment = Q tile rows ----
  const _Float16* qrow = qh + (size_t)(b * SEQQ + qi0 + fr) * DMODEL + h * HDIM;
  const v16h aq = cat16(*(const v8h*)(qrow + khalf), *(const v8h*)(qrow + 16 + khalf));

#pragma unroll
  for (int it = 0; it < 8; ++it) {
    const int kt  = wave + it * 8;                  // key tile 0..63
    const int key = kt * 16 + fr;
    const _Float16* krow = kh + (size_t)(b * SEQK + key) * DMODEL + h * HDIM;
    const v16h bk = cat16(*(const v8h*)(krow + khalf), *(const v8h*)(krow + 16 + khalf));
    v8f s = {};
    s = __builtin_amdgcn_wmma_f32_16x16x32_f16(false, aq, false, bk, (short)0, s, false, false);
    const int col = kt * 16 + fr;
#pragma unroll
    for (int r = 0; r < 8; ++r) Ss[(size_t)(rsel + r) * SEQK + col] = s[r] * scale;
  }
  __syncthreads();

  // ---- exact entmax15 per row ----
  for (int row = 0; row < 16; ++row) {
    float* xb = Ss + (size_t)row * SEQK;

    float lm = -3.4e38f;
    for (int t = tid; t < SEQK; t += 256) lm = fmaxf(lm, xb[t]);
    red1[tid] = lm;
    __syncthreads();
    for (int off = 128; off > 0; off >>= 1) {
      if (tid < off) red1[tid] = fmaxf(red1[tid], red1[tid + off]);
      __syncthreads();
    }
    const float xmax = red1[0];
    __syncthreads();
    for (int t = tid; t < SEQK; t += 256) {
      float xv = xb[t] - xmax;
      xb[t] = xv;
      sbuf[t] = xv;
    }
    __syncthreads();

    // bitonic ascending (read reversed => descending z)
    for (int ksz = 2; ksz <= SEQK; ksz <<= 1) {
      for (int j = ksz >> 1; j > 0; j >>= 1) {
        for (int i = tid; i < SEQK; i += 256) {
          int ixj = i ^ j;
          if (ixj > i) {
            float a = sbuf[i], c = sbuf[ixj];
            bool up = ((i & ksz) == 0);
            if ((a > c) == up) { sbuf[i] = c; sbuf[ixj] = a; }
          }
        }
        __syncthreads();
      }
    }

    // two-level inclusive prefix sums of z and z^2
    const int base = tid * 4;
    {
      float a1 = 0.f, a2 = 0.f;
#pragma unroll
      for (int u = 0; u < 4; ++u) {
        float z = sbuf[SEQK - 1 - (base + u)];
        a1 += z; a2 += z * z;
        cs1[base + u] = a1;
        cs2[base + u] = a2;
      }
      red1[tid] = a1;
      red2[tid] = a2;
    }
    __syncthreads();
    for (int off = 1; off < 256; off <<= 1) {
      float p1 = (tid >= off) ? red1[tid - off] : 0.f;
      float p2 = (tid >= off) ? red2[tid - off] : 0.f;
      __syncthreads();
      red1[tid] += p1;
      red2[tid] += p2;
      __syncthreads();
    }
    const float off1 = (tid > 0) ? red1[tid - 1] : 0.f;
    const float off2 = (tid > 0) ? red2[tid - 1] : 0.f;

    int cnt = 0;
#pragma unroll
    for (int u = 0; u < 4; ++u) {
      int j = base + u;
      float c1 = cs1[j] + off1, c2 = cs2[j] + off2;
      float kk = (float)(j + 1);
      float mean = c1 / kk, msq = c2 / kk;
      float ss = kk * (msq - mean * mean);
      float delta = fmaxf((1.f - ss) / kk, 0.f);
      float tau = mean - sqrtf(delta);
      taub[j] = tau;
      float z = sbuf[SEQK - 1 - j];
      if (tau <= z) cnt++;
    }
    __syncthreads();
    red1[tid] = (float)cnt;
    __syncthreads();
    for (int off = 128; off > 0; off >>= 1) {
      if (tid < off) red1[tid] += red1[tid + off];
      __syncthreads();
    }
    if (tid == 0) {
      int sup = (int)(red1[0] + 0.5f);
      s_support = sup < 1 ? 1 : sup;
    }
    __syncthreads();
    const float tau_star = taub[s_support - 1];

    for (int t = tid; t < SEQK; t += 256) {
      float p = fmaxf(xb[t] - tau_star, 0.f);
      xb[t] = p * p;
    }
    __syncthreads();
  }

  // ---- out = attn @ V : thread = (dim, row-pair); one V stream, two rows ----
  {
    const int d  = tid & (HDIM - 1);
    const int rp = tid >> 5;                         // 0..7 -> rows rp, rp+8
    const float* vbase = v + (size_t)(b * SEQK) * DMODEL + h * HDIM + d;
    const float* p0 = Ss + (size_t)rp * SEQK;
    const float* p1 = Ss + (size_t)(rp + 8) * SEQK;
    float s0 = 0.f, s1 = 0.f;
    for (int t = 0; t < SEQK; ++t) {
      const float vv = vbase[(size_t)t * DMODEL];
      s0 += p0[t] * vv;
      s1 += p1[t] * vv;
    }
    out[(size_t)(b * SEQQ + qi0 + rp) * DMODEL + h * HDIM + d]     = (_Float16)s0;
    out[(size_t)(b * SEQQ + qi0 + rp + 8) * DMODEL + h * HDIM + d] = (_Float16)s1;
  }
}

// ---------------------------------------------------------------------------
__global__ __launch_bounds__(256) void copy4_kernel(const float* __restrict__ src,
                                                    float* __restrict__ dst, int n4) {
  int i = blockIdx.x * 256 + threadIdx.x;
  if (i < n4) ((float4*)dst)[i] = ((const float4*)src)[i];
}

// ---------------------------------------------------------------------------
extern "C" void kernel_launch(void* const* d_in, const int* in_sizes, int n_in,
                              void* d_out, int out_size, void* d_ws, size_t ws_size,
                              hipStream_t stream)
{
  (void)in_sizes; (void)n_in; (void)out_size; (void)ws_size;

  const float* pose = (const float*)d_in[0];
  const float* vel  = (const float*)d_in[1];
  const float* img  = (const float*)d_in[2];
  const float* Wq   = (const float*)d_in[3];
  const float* bq   = (const float*)d_in[4];
  const float* Wk   = (const float*)d_in[5];
  const float* bk   = (const float*)d_in[6];
  const float* Wv   = (const float*)d_in[7];
  const float* bv   = (const float*)d_in[8];
  const float* Wo   = (const float*)d_in[9];
  const float* bo   = (const float*)d_in[10];
  const float* an_g = (const float*)d_in[11];
  const float* an_b = (const float*)d_in[12];
  const float* W1   = (const float*)d_in[13];
  const float* b1   = (const float*)d_in[14];
  const float* W2   = (const float*)d_in[15];
  const float* b2   = (const float*)d_in[16];
  const float* fn_g = (const float*)d_in[17];
  const float* fn_b = (const float*)d_in[18];
  const float* n_g  = (const float*)d_in[19];
  const float* n_b  = (const float*)d_in[20];

  const size_t TOK  = (size_t)NTOK * DMODEL;              // 524288
  const size_t WDD  = (size_t)LAYERS * DMODEL * DMODEL;   // 262144
  const size_t WDF  = (size_t)LAYERS * DMODEL * FFDIM;    // 786432
  const size_t H1N  = (size_t)NTOK * FFDIM;               // 1572864

  float* ws  = (float*)d_ws;
  float* X   = ws;            // f32 stream
  float* MEM = X   + TOK;
  float* Vb  = MEM + TOK;
  float* Ob  = Vb  + TOK;

  _Float16* hb   = (_Float16*)(Ob + TOK);
  _Float16* Xh   = hb;            // f16 activations
  _Float16* KVh  = Xh   + TOK;
  _Float16* Qh   = KVh  + TOK;
  _Float16* Kh   = Qh   + TOK;
  _Float16* ATTh = Kh   + TOK;
  _Float16* Ybh  = ATTh + TOK;
  _Float16* H1h  = Ybh  + TOK;
  _Float16* Wqh  = H1h  + H1N;    // f16 weights
  _Float16* Wkh  = Wqh  + WDD;
  _Float16* Wvh  = Wkh  + WDD;
  _Float16* Woh  = Wvh  + WDD;
  _Float16* W1h  = Woh  + WDD;
  _Float16* W2h  = W1h  + WDF;

  const int ATTN_LDS = (16 * SEQK + 4 * SEQK + 2 * 256) * (int)sizeof(float); // 83968 B

  dim3 blk(256);
  auto cvt = [&](const float* s, _Float16* d, size_t n) {
    int n4 = (int)(n / 4);
    cvt_f16_kernel<<<(n4 + 255) / 256, blk, 0, stream>>>(s, d, n4);
  };

  // Convert all weights to f16 once per launch.
  cvt(Wq, Wqh, WDD); cvt(Wk, Wkh, WDD); cvt(Wv, Wvh, WDD);
  cvt(Wo, Woh, WDD); cvt(W1, W1h, WDF); cvt(W2, W2h, WDF);

  auto run_stack = [&](const float* kv, float* dst) {
    copy4_kernel<<<(int)(TOK / 1024), blk, 0, stream>>>(pose, X, (int)(TOK / 4));
    cvt(kv, KVh, TOK);
    for (int i = 0; i < LAYERS; ++i) {
      copy4_kernel<<<(int)(TOK / 1024), blk, 0, stream>>>(X, MEM, (int)(TOK / 4));
      cvt(X, Xh, TOK);
      // --- MHA: Q,K projected straight to f16; V to f32 (for attn@V) ---
      gemm_wmma_kernel<DMODEL><<<dim3(NTOK / 128, DMODEL / 64), blk, 0, stream>>>(
          Xh,  Wqh + i * DMODEL * DMODEL, bq + i * DMODEL, nullptr, nullptr, Qh, DMODEL, ACT_NONE);
      gemm_wmma_kernel<DMODEL><<<dim3(NTOK / 128, DMODEL / 64), blk, 0, stream>>>(
          KVh, Wkh + i * DMODEL * DMODEL, bk + i * DMODEL, nullptr, nullptr, Kh, DMODEL, ACT_NONE);
      gemm_wmma_kernel<DMODEL><<<dim3(NTOK / 128, DMODEL / 64), blk, 0, stream>>>(
          KVh, Wvh + i * DMODEL * DMODEL, bv + i * DMODEL, nullptr, Vb, nullptr, DMODEL, ACT_NONE);
      attn_entmax_wmma_kernel<<<BATCH * NHEAD * (SEQQ / 16), blk, ATTN_LDS, stream>>>(
          Qh, Kh, Vb, ATTh);
      gemm_wmma_kernel<DMODEL><<<dim3(NTOK / 128, DMODEL / 64), blk, 0, stream>>>(
          ATTh, Woh + i * DMODEL * DMODEL, bo + i * DMODEL, nullptr, Ob, nullptr, DMODEL, ACT_NONE);
      // x = x + ln(attn_out)
      ln_add_kernel<<<NTOK, blk, 0, stream>>>(Ob, an_g + i * DMODEL, an_b + i * DMODEL, X, X, nullptr);
      // y = ln(x) -> f16 (feeds FFN GEMM)
      ln_add_kernel<<<NTOK, blk, 0, stream>>>(X, fn_g + i * DMODEL, fn_b + i * DMODEL, nullptr, nullptr, Ybh);
      // h = mish(y @ W1 + b1) -> f16 ; x = x + h @ W2 + b2
      gemm_wmma_kernel<DMODEL><<<dim3(NTOK / 128, FFDIM / 64), blk, 0, stream>>>(
          Ybh, W1h + i * DMODEL * FFDIM, b1 + i * FFDIM, nullptr, nullptr, H1h, FFDIM, ACT_MISH);
      gemm_wmma_kernel<FFDIM><<<dim3(NTOK / 128, DMODEL / 64), blk, 0, stream>>>(
          H1h, W2h + i * FFDIM * DMODEL, b2 + i * DMODEL, X, X, nullptr, DMODEL, ACT_NONE);
      // x = mem + ln(x)
      ln_add_kernel<<<NTOK, blk, 0, stream>>>(X, n_g + i * DMODEL, n_b + i * DMODEL, MEM, X, nullptr);
    }
    copy4_kernel<<<(int)(TOK / 1024), blk, 0, stream>>>(X, dst, (int)(TOK / 4));
  };

  float* outp = (float*)d_out;
  run_stack(img, outp + TOK);   // img_out (second tuple element)
  run_stack(vel, outp);         // pose_out (first tuple element)
}
